// SparseMoE_89498528514678
// MI455X (gfx1250) — matrix operand order repaired and checked by reading, not compiled, not run on hardware
//
#include <hip/hip_runtime.h>
#include <math.h>

// ---------------- types ----------------
typedef __attribute__((ext_vector_type(16))) _Float16 v16h;
typedef __attribute__((ext_vector_type(8)))  _Float16 v8h;
typedef __attribute__((ext_vector_type(8)))  float    v8f;
typedef __attribute__((ext_vector_type(4)))  int      v4i;

#define AS1 __attribute__((address_space(1)))
#define AS3 __attribute__((address_space(3)))

#if __has_builtin(__builtin_amdgcn_global_load_async_to_lds_b128)
#define HAVE_ASYNC_LDS 1
#else
#define HAVE_ASYNC_LDS 0
#endif

static __device__ __forceinline__ void wait_asynccnt0() {
#if __has_builtin(__builtin_amdgcn_s_wait_asynccnt)
  __builtin_amdgcn_s_wait_asynccnt(0);
#else
  asm volatile("s_wait_asynccnt 0x0" ::: "memory");
#endif
}

// ---------------- problem constants (fixed by harness shapes) ----------------
static constexpr int NTOK = 4096;   // B*T
static constexpr int CC   = 1024;   // C
static constexpr int HH   = 4096;   // 4C
static constexpr int C2   = 2048;   // 2C
static constexpr int EE   = 6;
static constexpr float TEMP = 0.95f; // 0.95**(1024//1000), clipped to [0.1,1]

// ---------------- GEMM tiling ----------------
static constexpr int BM = 128, BN = 64, BK = 32;
static constexpr int LDA_S = 40;   // padded f16 row stride (80B) -> bank-conflict-free b128 frag loads
static constexpr int LDB_S = 40;

__device__ __forceinline__ float act_apply(int ACT, float v) {
  if (ACT == 1) return 0.5f * v * (1.0f + erff(v * 0.70710678118654752f)); // exact GELU
  if (ACT == 2) return v / (1.0f + expf(-v));                              // SiLU
  return v;
}

// C[M,N] = act(A[M,K] @ B[K,N] + bias[N]); A,B f16 row-major packed; out f32 or f16 (ldc-strided)
template<int ACT, bool OUTH>
__global__ __launch_bounds__(256)
void gemm_wmma(const _Float16* __restrict__ A, const _Float16* __restrict__ B,
               const float* __restrict__ bias,
               float* __restrict__ Cf, _Float16* __restrict__ Ch,
               int N, int K, int ldc)
{
  __shared__ alignas(16) _Float16 As[BM * LDA_S];
  __shared__ alignas(16) _Float16 Bs[BN * LDB_S];

  const int tid  = threadIdx.x;
  const int lane = tid & 31, wave = tid >> 5;
  const int wm = wave & 3, wn = wave >> 2;         // 4x2 wave grid -> 128x64
  const int half16 = lane >> 4, l16 = lane & 15;
  const int m0 = blockIdx.y * BM, n0 = blockIdx.x * BN;

  // cooperative load coordinates
  const int arow = tid >> 2;          // 0..63 (two rows per thread cover 128)
  const int akq  = (tid & 3) * 8;     // 0,8,16,24
  const int brow = tid >> 3;          // 0..31 (k)
  const int bn8  = (tid & 7) * 8;     // 0..56 (n)

  // hoisted global/LDS pointers (stride by BK / BK*N per K-tile)
  const _Float16* pA0 = A + (size_t)(m0 + arow)      * K + akq;
  const _Float16* pA1 = A + (size_t)(m0 + 64 + arow) * K + akq;
  const _Float16* pB  = B + (size_t)brow * N + n0 + bn8;
  _Float16* lA0 = As + arow * LDA_S + akq;
  _Float16* lA1 = As + (64 + arow) * LDA_S + akq;

  v8f acc[2][2] = {};
  const int kt_count = K / BK;

  for (int kt = 0; kt < kt_count; ++kt) {
    v8h b0 = *(const v8h*)pB;
    if (kt + 1 < kt_count)              // global_prefetch_b8 for next B K-tile
      __builtin_prefetch(pB + (size_t)BK * N, 0, 1);
#if !HAVE_ASYNC_LDS
    v8h a0 = *(const v8h*)pA0;
    v8h a1 = *(const v8h*)pA1;
#endif
    __syncthreads();                    // previous iteration's compute done
#if HAVE_ASYNC_LDS
    // A tile: global -> LDS via ASYNCcnt-tracked async copy (no VGPR staging)
    __builtin_amdgcn_global_load_async_to_lds_b128(
        (AS1 v4i*)pA0, (AS3 v4i*)lA0, 0, 0);
    __builtin_amdgcn_global_load_async_to_lds_b128(
        (AS1 v4i*)pA1, (AS3 v4i*)lA1, 0, 0);
#else
    *(v8h*)lA0 = a0;
    *(v8h*)lA1 = a1;
#endif
#pragma unroll
    for (int j = 0; j < 8; ++j) Bs[(bn8 + j) * LDB_S + brow] = b0[j]; // transpose to [n][k]
#if HAVE_ASYNC_LDS
    wait_asynccnt0();
#endif
    __syncthreads();

    // A fragment: row m=l16, K set {k0..k0+7, k0+16..k0+23}, k0 = half16*8   (ISA 7.12.2)
    v16h af[2], bf[2];
    const int ak0 = half16 * 8;
#pragma unroll
    for (int im = 0; im < 2; ++im) {
      const _Float16* p = As + (wm * 32 + im * 16 + l16) * LDA_S + ak0;
      v8h lo = *(const v8h*)p;
      v8h hi = *(const v8h*)(p + 16);
#pragma unroll
      for (int j = 0; j < 8; ++j) { af[im][j] = lo[j]; af[im][8 + j] = hi[j]; }
    }
    // B fragment: col n=l16, K set {k0..k0+7, k0+16..k0+23}, k0 = half16*8
    const int bk0 = half16 * 8;
#pragma unroll
    for (int in = 0; in < 2; ++in) {
      const _Float16* p = Bs + (wn * 32 + in * 16 + l16) * LDB_S + bk0;
      v8h lo = *(const v8h*)p;
      v8h hi = *(const v8h*)(p + 16);
#pragma unroll
      for (int j = 0; j < 8; ++j) { bf[in][j] = lo[j]; bf[in][8 + j] = hi[j]; }
    }
#pragma unroll
    for (int im = 0; im < 2; ++im)
#pragma unroll
      for (int in = 0; in < 2; ++in)
        acc[im][in] = __builtin_amdgcn_wmma_f32_16x16x32_f16(
            false, af[im], false, bf[in], (short)0, acc[im][in], false, false);

    pA0 += BK; pA1 += BK; pB += (size_t)BK * N;
  }

  // epilogue: VGPR j -> M = j + 8*half16; N = l16   (ISA 7.12.2 C/D layout)
#pragma unroll
  for (int im = 0; im < 2; ++im) {
#pragma unroll
    for (int in = 0; in < 2; ++in) {
      const int col = n0 + wn * 32 + in * 16 + l16;
      const float bv = bias[col];
#pragma unroll
      for (int j = 0; j < 8; ++j) {
        const int row = m0 + wm * 32 + im * 16 + half16 * 8 + j;
        float v = act_apply(ACT, acc[im][in][j] + bv);
        if (OUTH) Ch[(size_t)row * ldc + col] = (_Float16)v;
        else      Cf[(size_t)row * ldc + col] = v;
      }
    }
  }
}

// ---------------- reductions ----------------
template<int NT>
__device__ __forceinline__ float blockReduceSum(float v, float* red) {
  constexpr int NW = NT / 32;
  for (int o = 16; o > 0; o >>= 1) v += __shfl_down(v, o, 32);
  const int wid = threadIdx.x >> 5, ln = threadIdx.x & 31;
  if (ln == 0) red[wid] = v;
  __syncthreads();
  float t = (threadIdx.x < NW) ? red[threadIdx.x] : 0.0f;
  if (wid == 0) {
    for (int o = NW >> 1; o > 0; o >>= 1) t += __shfl_down(t, o, 32);
    if (ln == 0) red[0] = t;
  }
  __syncthreads();
  float r = red[0];
  __syncthreads();
  return r;
}

// ---------------- LayerNorm (one block per row) ----------------
template<int ACT, bool RES, bool OUTH>
__global__ __launch_bounds__(256)
void ln_rows(const float* __restrict__ in, const float* __restrict__ resid,
             const float* __restrict__ g, const float* __restrict__ b,
             float* __restrict__ outF, _Float16* __restrict__ outH, int D)
{
  __shared__ float red[8];
  const size_t base = (size_t)blockIdx.x * D;
  float s = 0.f, s2 = 0.f;
  for (int c = threadIdx.x; c < D; c += 256) {
    float z = in[base + c] + (RES ? resid[base + c] : 0.0f);
    s += z; s2 += z * z;
  }
  s  = blockReduceSum<256>(s,  red);
  s2 = blockReduceSum<256>(s2, red);
  const float mean = s / D;
  const float inv  = rsqrtf(s2 / D - mean * mean + 1e-5f);
  for (int c = threadIdx.x; c < D; c += 256) {
    float z = in[base + c] + (RES ? resid[base + c] : 0.0f);
    float y = act_apply(ACT, (z - mean) * inv * g[c] + b[c]);
    if (OUTH) outH[base + c] = (_Float16)y;
    else      outF[base + c] = y;
  }
}

// ---------------- misc elementwise ----------------
__global__ __launch_bounds__(256)
void cvt_f2h(const float* __restrict__ s, _Float16* __restrict__ d, size_t n) {
  size_t i = (size_t)blockIdx.x * 256 + threadIdx.x;
  if (i < n) d[i] = (_Float16)s[i];
}

__global__ __launch_bounds__(256)
void tanorm_kernel(const float* __restrict__ ta, float* __restrict__ out, int C) {
  __shared__ float red[8];
  float s = 0.f;
  for (int c = threadIdx.x; c < C; c += 256) {
    float v = ta[(size_t)blockIdx.x * C + c];
    s += v * v;
  }
  s = blockReduceSum<256>(s, red);
  if (threadIdx.x == 0) out[blockIdx.x] = fmaxf(sqrtf(s), 1e-8f);
}

// ---------------- router + noisy top-2 gating + type-aware scale ----------------
__global__ __launch_bounds__(128)
void router_kernel(const float* __restrict__ x, const float* __restrict__ q,
                   const float* __restrict__ tq, const float* __restrict__ ta,
                   const float* __restrict__ tanorm,
                   const float* __restrict__ nw, const float* __restrict__ nb,
                   const float* __restrict__ noise, float* __restrict__ coef, int C)
{
  __shared__ float red[4];
  const int n = blockIdx.x;
  const float* px = x + (size_t)n * C;
  const float* pq = q + (size_t)n * C;

  float part[14];
#pragma unroll
  for (int i = 0; i < 14; ++i) part[i] = 0.f;
  for (int c = threadIdx.x; c < C; c += 128) {
    const float xv = px[c], qv = pq[c];
    part[12] += xv * nw[c];
    part[13] += xv * xv;
#pragma unroll
    for (int e = 0; e < EE; ++e) {
      part[e]      += qv * tq[e * C + c];
      part[6 + e]  += xv * ta[e * C + c];
    }
  }
  float S[14];
  for (int i = 0; i < 14; ++i) S[i] = blockReduceSum<128>(part[i], red);

  if (threadIdx.x == 0) {
    const float gn   = 1.0f / (1.0f + expf(-(S[12] + nb[0])));
    const float xnrm = fmaxf(sqrtf(S[13]), 1e-8f);
    float noisy[EE], scale[EE];
#pragma unroll
    for (int e = 0; e < EE; ++e) {
      noisy[e] = S[e] + TEMP * noise[(size_t)n * EE + e] * gn;
      scale[e] = 1.0f + 0.2f * S[6 + e] / (xnrm * tanorm[e]);
    }
    int i1 = 0;
    for (int e = 1; e < EE; ++e) if (noisy[e] > noisy[i1]) i1 = e;
    int i2 = (i1 == 0) ? 1 : 0;
    for (int e = 0; e < EE; ++e) if (e != i1 && noisy[e] > noisy[i2]) i2 = e;
    const float it = 1.0f / (TEMP + 1e-6f);
    const float m  = fmaxf(noisy[i1], noisy[i2]);
    const float e1 = expf((noisy[i1] - m) * it), e2 = expf((noisy[i2] - m) * it);
    const float inv = 1.0f / (e1 + e2);
#pragma unroll
    for (int e = 0; e < EE; ++e) coef[(size_t)n * EE + e] = 0.0f;
    coef[(size_t)n * EE + i1] = e1 * inv * scale[i1];
    coef[(size_t)n * EE + i2] = e2 * inv * scale[i2];
  }
}

// ---------------- combine: out[n,c] = sum_e coef[n,e] * eo[e][n,c] ----------------
__global__ __launch_bounds__(256)
void combine_kernel(const float* __restrict__ eo, const float* __restrict__ coef,
                    float* __restrict__ out, int total, int C, size_t strideE)
{
  int idx = blockIdx.x * 256 + threadIdx.x;
  if (idx >= total) return;
  const int n = idx / C;
  float s = 0.f;
#pragma unroll
  for (int e = 0; e < EE; ++e)
    s += coef[(size_t)n * EE + e] * eo[(size_t)e * strideE + idx];
  out[idx] = s;
}

// ---------------- host-side launch helpers ----------------
template<int ACT, bool OUTH>
static void launch_gemm(const _Float16* A, const _Float16* B, const float* bias,
                        float* Cf, _Float16* Ch, int M, int N, int K, int ldc,
                        hipStream_t s)
{
  dim3 grid(N / BN, M / BM);
  gemm_wmma<ACT, OUTH><<<grid, 256, 0, s>>>(A, B, bias, Cf, Ch, N, K, ldc);
}

extern "C" void kernel_launch(void* const* d_in, const int* in_sizes, int n_in,
                              void* d_out, int out_size, void* d_ws, size_t ws_size,
                              hipStream_t stream)
{
  (void)in_sizes; (void)n_in; (void)out_size; (void)ws_size;
  // inputs in setup_inputs() order
  const float* x    = (const float*)d_in[0];
  const float* noise= (const float*)d_in[1];
  const float* rw1  = (const float*)d_in[2];
  const float* rb1  = (const float*)d_in[3];
  const float* rw2  = (const float*)d_in[4];
  const float* rb2  = (const float*)d_in[5];
  const float* nw   = (const float*)d_in[6];
  const float* nb   = (const float*)d_in[7];
  const float* tq   = (const float*)d_in[8];
  const float* ta   = (const float*)d_in[9];
  const float* d_w1 = (const float*)d_in[10];
  const float* d_b1 = (const float*)d_in[11];
  const float* d_w2 = (const float*)d_in[12];
  const float* d_b2 = (const float*)d_in[13];
  const float* d_lg = (const float*)d_in[14];
  const float* d_lb = (const float*)d_in[15];
  const float* d_w3 = (const float*)d_in[16];
  const float* d_b3 = (const float*)d_in[17];
  const float* d_ng = (const float*)d_in[18];
  const float* d_nb = (const float*)d_in[19];
  const float* w_w1 = (const float*)d_in[20];
  const float* w_b1 = (const float*)d_in[21];
  const float* w_lg = (const float*)d_in[22];
  const float* w_lb = (const float*)d_in[23];
  const float* w_w2 = (const float*)d_in[24];
  const float* w_b2 = (const float*)d_in[25];
  const float* w_ng = (const float*)d_in[26];
  const float* w_nb = (const float*)d_in[27];
  const float* h_a1 = (const float*)d_in[28];
  const float* h_ab1= (const float*)d_in[29];
  const float* h_a2 = (const float*)d_in[30];
  const float* h_ab2= (const float*)d_in[31];
  const float* h_b1 = (const float*)d_in[32];
  const float* h_bb1= (const float*)d_in[33];
  const float* h_b2 = (const float*)d_in[34];
  const float* h_bb2= (const float*)d_in[35];
  const float* h_pw = (const float*)d_in[36];
  const float* h_pb = (const float*)d_in[37];
  const float* h_ng = (const float*)d_in[38];
  const float* h_nb = (const float*)d_in[39];
  float* out = (float*)d_out;

  const int Nt = NTOK, C = CC, H = HH;

  // workspace carve-out (~480MB)
  char* wsp = (char*)d_ws;
  auto alloc = [&](size_t bytes) -> char* {
    char* p = wsp; wsp += (bytes + 255) & ~(size_t)255; return p;
  };
  _Float16* xh    = (_Float16*)alloc((size_t)Nt * C * 2);
  _Float16* rw1h  = (_Float16*)alloc((size_t)C * H * 2);
  _Float16* rw2h  = (_Float16*)alloc((size_t)H * C * 2);
  _Float16* dw1h  = (_Float16*)alloc((size_t)2 * C * H * 2);
  _Float16* dw2h  = (_Float16*)alloc((size_t)2 * H * H * 2);
  _Float16* dw3h  = (_Float16*)alloc((size_t)2 * H * C * 2);
  _Float16* ww1h  = (_Float16*)alloc((size_t)2 * C * H * 2);
  _Float16* ww2h  = (_Float16*)alloc((size_t)2 * H * C * 2);
  _Float16* ha1h  = (_Float16*)alloc((size_t)2 * C * H * 2);
  _Float16* ha2h  = (_Float16*)alloc((size_t)2 * H * C * 2);
  _Float16* hb1h  = (_Float16*)alloc((size_t)2 * C * C2 * 2);
  _Float16* hb2h  = (_Float16*)alloc((size_t)2 * C2 * C * 2);
  _Float16* hpwh  = (_Float16*)alloc((size_t)2 * 2 * C * C * 2);
  _Float16* t1h   = (_Float16*)alloc((size_t)Nt * H * 2);
  _Float16* t2h   = (_Float16*)alloc((size_t)Nt * H * 2);
  _Float16* combh = (_Float16*)alloc((size_t)Nt * 2 * C * 2);
  float*    qf    = (float*)alloc((size_t)Nt * C * 4);
  float*    midf  = (float*)alloc((size_t)Nt * H * 4);
  float*    yf    = (float*)alloc((size_t)Nt * C * 4);
  float*    eo    = (float*)alloc((size_t)EE * Nt * C * 4);
  float*    coef  = (float*)alloc((size_t)Nt * EE * 4);
  float*    tanrm = (float*)alloc((size_t)EE * 4);

  auto cvt = [&](const float* s, _Float16* d, size_t n) {
    cvt_f2h<<<dim3((unsigned)((n + 255) / 256)), 256, 0, stream>>>(s, d, n);
  };

  // fp32 -> f16 weight/activation staging
  cvt(x,    xh,   (size_t)Nt * C);
  cvt(rw1,  rw1h, (size_t)C * H);
  cvt(rw2,  rw2h, (size_t)H * C);
  cvt(d_w1, dw1h, (size_t)2 * C * H);
  cvt(d_w2, dw2h, (size_t)2 * H * H);
  cvt(d_w3, dw3h, (size_t)2 * H * C);
  cvt(w_w1, ww1h, (size_t)2 * C * H);
  cvt(w_w2, ww2h, (size_t)2 * H * C);
  cvt(h_a1, ha1h, (size_t)2 * C * H);
  cvt(h_a2, ha2h, (size_t)2 * H * C);
  cvt(h_b1, hb1h, (size_t)2 * C * C2);
  cvt(h_b2, hb2h, (size_t)2 * C2 * C);
  cvt(h_pw, hpwh, (size_t)2 * 2 * C * C);

  tanorm_kernel<<<EE, 256, 0, stream>>>(ta, tanrm, C);

  // ---- router: q = gelu(x@rw1+b1)@rw2+b2, then noisy top-2 gating ----
  launch_gemm<1, true >(xh,  rw1h, rb1, nullptr, t1h, Nt, H, C, H, stream);
  launch_gemm<0, false>(t1h, rw2h, rb2, qf, nullptr, Nt, C, H, C, stream);
  router_kernel<<<Nt, 128, 0, stream>>>(x, qf, tq, ta, tanrm, nw, nb, noise, coef, C);

  // ---- deep experts: silu(x@w1)@w2 -> LN -> silu -> @w3, residual+LN ----
  for (int k = 0; k < 2; ++k) {
    launch_gemm<2, true >(xh,  dw1h + (size_t)k * C * H, d_b1 + k * H, nullptr, t1h, Nt, H, C, H, stream);
    launch_gemm<0, false>(t1h, dw2h + (size_t)k * H * H, d_b2 + k * H, midf, nullptr, Nt, H, H, H, stream);
    ln_rows<2, false, true ><<<Nt, 256, 0, stream>>>(midf, nullptr, d_lg + k * H, d_lb + k * H, nullptr, t2h, H);
    launch_gemm<0, false>(t2h, dw3h + (size_t)k * H * C, d_b3 + k * C, yf, nullptr, Nt, C, H, C, stream);
    ln_rows<0, true , false><<<Nt, 256, 0, stream>>>(yf, x, d_ng + k * C, d_nb + k * C, eo + (size_t)k * Nt * C, nullptr, C);
  }

  // ---- wide experts: LN(gelu(x@w1)) @ w2, residual+LN ----
  for (int k = 0; k < 2; ++k) {
    launch_gemm<1, false>(xh, ww1h + (size_t)k * C * H, w_b1 + k * H, midf, nullptr, Nt, H, C, H, stream);
    ln_rows<0, false, true ><<<Nt, 256, 0, stream>>>(midf, nullptr, w_lg + k * H, w_lb + k * H, nullptr, t2h, H);
    launch_gemm<0, false>(t2h, ww2h + (size_t)k * H * C, w_b2 + k * C, yf, nullptr, Nt, C, H, C, stream);
    ln_rows<0, true , false><<<Nt, 256, 0, stream>>>(yf, x, w_ng + k * C, w_nb + k * C, eo + (size_t)(2 + k) * Nt * C, nullptr, C);
  }

  // ---- hybrid experts: proj(cat(gelu-path, silu-path)), residual+LN ----
  for (int k = 0; k < 2; ++k) {
    launch_gemm<1, true >(xh,  ha1h + (size_t)k * C * H,  h_ab1 + k * H,  nullptr, t1h,       Nt, H,  C,  H,     stream);
    launch_gemm<0, true >(t1h, ha2h + (size_t)k * H * C,  h_ab2 + k * C,  nullptr, combh,     Nt, C,  H,  2 * C, stream);
    launch_gemm<2, true >(xh,  hb1h + (size_t)k * C * C2, h_bb1 + k * C2, nullptr, t1h,       Nt, C2, C,  C2,    stream);
    launch_gemm<0, true >(t1h, hb2h + (size_t)k * C2 * C, h_bb2 + k * C,  nullptr, combh + C, Nt, C,  C2, 2 * C, stream);
    launch_gemm<0, false>(combh, hpwh + (size_t)k * 2 * C * C, h_pb + k * C, yf, nullptr, Nt, C, 2 * C, C, stream);
    ln_rows<0, true, false><<<Nt, 256, 0, stream>>>(yf, x, h_ng + k * C, h_nb + k * C, eo + (size_t)(4 + k) * Nt * C, nullptr, C);
  }

  // ---- gated combine ----
  const int total = Nt * C;
  combine_kernel<<<(total + 255) / 256, 256, 0, stream>>>(eo, coef, out, total, C, (size_t)Nt * C);
}
